// SVDHead_29678224015624
// MI455X (gfx1250) — matrix operand order repaired and checked by reading, not compiled, not run on hardware
//
#include <hip/hip_runtime.h>
#include <hip/hip_bf16.h>

// ---------------- problem constants (match reference setup_inputs) ---------
#define Bh   4
#define Nn   768
#define Mm   768
#define Cc   128
#define Kk   16      // K_NN
#define K1n  8       // K1_NN
#define NKP  256     // NUM_KEYPOINTS
#define BNM  (Bh*Nn*Mm)

// output layout (floats, concatenated in return order)
#define OUT_R       0        // B*3*3 = 36
#define OUT_T       36       // B*3   = 12
#define OUT_SKP     48       // B*3*256 = 3072
#define OUT_TKP     3120     // 3072
#define OUT_SKNN    6192     // B*3*256*16 = 49152
#define OUT_TKNN    55344    // 49152
#define OUT_LOSS    104496   // 1

// workspace layout (float offsets)
#define WS_XX    0u
#define WS_YY    3072u
#define WS_DIST  6144u
#define WS_SCORE (WS_DIST + (unsigned)BNM)
#define WS_A     (WS_SCORE + (unsigned)BNM)
#define WS_CORR  (WS_A + (unsigned)BNM)          // B*3*N = 9216
#define WS_W     (WS_CORR + 9216u)               // B*N = 3072
#define WS_HM    (WS_W + 3072u)                  // B*16 = 64
#define WS_TOPK  (WS_HM + 64u)                   // int: B*256
#define WS_ARG   (WS_TOPK + 1024u)               // int: B*N

typedef __attribute__((ext_vector_type(16))) _Float16 v16h;
typedef __attribute__((ext_vector_type(8)))  float    v8f;
typedef __attribute__((ext_vector_type(4)))  unsigned int u32x4;
typedef __attribute__((ext_vector_type(4)))  int      i32x4;
typedef __attribute__((ext_vector_type(8)))  int      i32x8;

union H16u { v16h v; _Float16 e[16]; };
union F8u  { v8f  v; float e[8]; };

#if __has_builtin(__builtin_amdgcn_tensor_load_to_lds)
#define HAVE_TDM 1
#else
#define HAVE_TDM 0
#endif

// ------------------------------ norms --------------------------------------
__global__ __launch_bounds__(256) void kn_norms(const float* __restrict__ xe,
                                                const float* __restrict__ ye,
                                                float* __restrict__ xx,
                                                float* __restrict__ yy) {
  int idx = blockIdx.x * blockDim.x + threadIdx.x;
  if (idx < Bh * Nn) {
    int b = idx / Nn, n = idx % Nn;
    float s = 0.f;
    for (int c = 0; c < Cc; ++c) { float v = xe[(b*Cc + c)*Nn + n]; s += v*v; }
    xx[idx] = s;
  } else if (idx < Bh*(Nn+Mm)) {
    int j = idx - Bh*Nn;
    int b = j / Mm, m = j % Mm;
    float s = 0.f;
    for (int c = 0; c < Cc; ++c) { float v = ye[(b*Cc + c)*Mm + m]; s += v*v; }
    yy[j] = s;
  }
}

// --------------------- TDM 2D tile load (global -> LDS) --------------------
#if HAVE_TDM
__device__ __forceinline__ void tdm_load_tile_2d(unsigned lds_byte_off,
                                                 const float* gptr,
                                                 unsigned tensor_w, unsigned tensor_h,
                                                 unsigned row_stride_elems,
                                                 unsigned tile_w, unsigned tile_h) {
  const unsigned long long ga = (unsigned long long)(uintptr_t)gptr;
  // D# group 0: count=1 (valid user descriptor), lds_addr, global_addr, type=2
  u32x4 g0;
  g0[0] = 1u;
  g0[1] = lds_byte_off;
  g0[2] = (unsigned)ga;
  g0[3] = (unsigned)((ga >> 32) & 0x01ffffffull) | (2u << 30);
  // D# group 1: data_size=4B (code 2), tensor dims, tile dims, dim0 stride
  i32x8 g1;
  g1[0] = (int)(2u << 16);                                   // data_size
  g1[1] = (int)((tensor_w & 0xffffu) << 16);                 // tensor_dim0 lo16
  g1[2] = (int)(((tensor_w >> 16) & 0xffffu) |
                ((tensor_h & 0xffffu) << 16));               // dim0 hi | dim1 lo
  g1[3] = (int)(((tensor_h >> 16) & 0xffffu) |
                ((tile_w & 0xffffu) << 16));                 // dim1 hi | tile0
  g1[4] = (int)(tile_h & 0xffffu);                           // tile1 (tile2=0)
  g1[5] = (int)row_stride_elems;                             // dim0_stride lo32
  g1[6] = 0;
  g1[7] = 0;
  i32x4 z4 = {0, 0, 0, 0};
#if __clang_major__ >= 23
  i32x8 z8 = {0, 0, 0, 0, 0, 0, 0, 0};
  __builtin_amdgcn_tensor_load_to_lds(g0, g1, z4, z4, z8, 0);
#else
  __builtin_amdgcn_tensor_load_to_lds(g0, g1, z4, z4, 0);
#endif
}
#endif

// ------------------------ pairwise distance via WMMA -----------------------
// dist[b,n,m] = xx[n] + yy[m] - 2 * sum_c xe[b,c,n]*ye[b,c,m]
// block (128 thr = 4 waves) computes 32n x 64m; A/B tiles staged in LDS
// (via TDM when available), each wave runs 2 f32 accumulators sharing B.
__global__ __launch_bounds__(128) void kn_dist_wmma(const float* __restrict__ xe,
                                                    const float* __restrict__ ye,
                                                    const float* __restrict__ xx,
                                                    const float* __restrict__ yy,
                                                    float* __restrict__ dist) {
  __shared__ __align__(16) float sA[Cc * 32];   // [c][n], 16 KB
  __shared__ __align__(16) float sB[Cc * 64];   // [c][m], 32 KB

  const int tid  = threadIdx.x;
  const int wave = tid >> 5;
  const int lane = tid & 31;
  const int g    = lane >> 4;
  const int l16  = lane & 15;

  const int n0   = blockIdx.x * 32;
  const int mblk = blockIdx.y * 64;
  const int b    = blockIdx.z;

  const float* X = xe + (size_t)b * Cc * Nn;    // (C,N)
  const float* Y = ye + (size_t)b * Cc * Mm;    // (C,M)

#if HAVE_TDM
  if (wave == 0) {   // one wave issues both DMA descriptors
    tdm_load_tile_2d((unsigned)(uintptr_t)(void*)sA, X + n0, Nn, Cc, Nn, 32, Cc);
    tdm_load_tile_2d((unsigned)(uintptr_t)(void*)sB, Y + mblk, Mm, Cc, Mm, 64, Cc);
    __builtin_amdgcn_s_wait_tensorcnt(0);
  }
  __syncthreads();
#else
  for (int idx = tid; idx < Cc * 8; idx += 128) {     // A: 8 float4 per row
    const int c = idx >> 3, q = idx & 7;
    ((float4*)sA)[c * 8 + q] = *(const float4*)(X + (size_t)c * Nn + n0 + q * 4);
  }
  for (int idx = tid; idx < Cc * 16; idx += 128) {    // B: 16 float4 per row
    const int c = idx >> 4, q = idx & 15;
    ((float4*)sB)[c * 16 + q] = *(const float4*)(Y + (size_t)c * Mm + mblk + q * 4);
  }
  __syncthreads();
#endif

  F8u acc0, acc1; acc0.v = {}; acc1.v = {};
  const int mloc = wave * 16 + l16;

#pragma unroll
  for (int kb = 0; kb < 4; ++kb) {
    const int c0 = kb * 32;
    H16u a0, a1, bb;
#pragma unroll
    for (int i = 0; i < 16; ++i) {
      const int r = i >> 1, sub = i & 1;
      // A (16x32 f16) operand layout: row=l16, K selected by (VGPR,half,group)
      const int ka  = ((r & 3) << 1) + sub + (g << 3) + ((r >> 2) << 4);
      // B (32x16 f16) operand layout: col=l16, K selected same as A
      const int kbv = ((r & 3) << 1) + sub + (g << 3) + ((r >> 2) << 4);
      a0.e[i] = (_Float16)sA[(c0 + ka) * 32 + l16];
      a1.e[i] = (_Float16)sA[(c0 + ka) * 32 + 16 + l16];
      bb.e[i] = (_Float16)sB[(c0 + kbv) * 64 + mloc];
    }
    acc0.v = __builtin_amdgcn_wmma_f32_16x16x32_f16(false, a0.v, false, bb.v,
                                                    (short)0, acc0.v, false, false);
    acc1.v = __builtin_amdgcn_wmma_f32_16x16x32_f16(false, a1.v, false, bb.v,
                                                    (short)0, acc1.v, false, false);
  }

  // C/D layout: lane 0-15 -> N=lane, VGPR j -> M=j; lanes 16-31 -> M=j+8
  const int m = mblk + wave * 16 + l16;
  const float yv = yy[b * Mm + m];
#pragma unroll
  for (int j = 0; j < 8; ++j) {
    int n = n0 + j + (g << 3);
    dist[((size_t)(b * Nn + n)) * Mm + m] = xx[b * Nn + n] + yv - 2.0f * acc0.e[j];
    n += 16;
    dist[((size_t)(b * Nn + n)) * Mm + m] = xx[b * Nn + n] + yv - 2.0f * acc1.e[j];
  }
}

// ----------------------- row softmax of (-in), fused argmax ----------------
__global__ __launch_bounds__(256) void kn_row_softmax(const float* __restrict__ in,
                                                      float* __restrict__ out,
                                                      int* __restrict__ argm) {
  const int row = blockIdx.x;              // b*N + n
  const int tid = threadIdx.x;
  const size_t base = (size_t)row * Mm;
  __shared__ float rmax[256];
  __shared__ int   rarg[256];

  float lmax = -3.4e38f; int lidx = 0;
  for (int m = tid; m < Mm; m += 256) {
    float v = -in[base + m];
    if (v > lmax) { lmax = v; lidx = m; }
  }
  rmax[tid] = lmax; rarg[tid] = lidx; __syncthreads();
  for (int s = 128; s > 0; s >>= 1) {
    if (tid < s) {
      float a = rmax[tid], b2 = rmax[tid + s];
      int ia = rarg[tid], ib = rarg[tid + s];
      if (b2 > a || (b2 == a && ib < ia)) { rmax[tid] = b2; rarg[tid] = ib; }
    }
    __syncthreads();
  }
  const float mx = rmax[0];
  const int best = rarg[0];
  __syncthreads();

  float lsum = 0.f;
  for (int m = tid; m < Mm; m += 256) lsum += expf(-in[base + m] - mx);
  rmax[tid] = lsum; __syncthreads();
  for (int s = 128; s > 0; s >>= 1) {
    if (tid < s) rmax[tid] += rmax[tid + s];
    __syncthreads();
  }
  const float inv = 1.0f / rmax[0];
  __syncthreads();

  for (int m = tid; m < Mm; m += 256)
    out[base + m] = expf(-in[base + m] - mx) * inv;   // safe in-place
  if (argm && tid == 0) argm[row] = best;
}

// -------- A[b,n,m] = sum_{j=1..7} scores[src_idx1[(b*N+n)*8+j]][m] ---------
__global__ __launch_bounds__(256) void kn_A(const float* __restrict__ scores,
                                            const int* __restrict__ src_idx1,
                                            float* __restrict__ A) {
  const int row = blockIdx.x;
  const int tid = threadIdx.x;
  __shared__ int ridx[K1n - 1];
  if (tid < K1n - 1) {
    const int r = src_idx1[row * K1n + 1 + tid];
    ridx[tid] = r;
    __builtin_prefetch(&scores[(size_t)r * Mm], 0, 1);   // global_prefetch_b8
  }
  __syncthreads();
  for (int m = tid; m < Mm; m += 256) {
    float s = 0.f;
#pragma unroll
    for (int j = 0; j < K1n - 1; ++j) s += scores[(size_t)ridx[j] * Mm + m];
    A[(size_t)row * Mm + m] = s;
  }
}

// refined = exp(margin - consensus) * dist ; consensus = mean_j A[b,n,idx2[b,m,1+j]]
__global__ __launch_bounds__(256) void kn_refined(const float* __restrict__ A,
                                                  const float* __restrict__ dist,
                                                  const int* __restrict__ idx2,
                                                  const float* __restrict__ margin_p,
                                                  float* __restrict__ refined) {
  const float margin = *margin_p;
  size_t i = (size_t)blockIdx.x * blockDim.x + threadIdx.x;
  if (i >= (size_t)BNM) return;
  const int b = (int)(i / ((size_t)Nn * Mm));
  const int rem = (int)(i - (size_t)b * Nn * Mm);
  const int n = rem / Mm, m = rem % Mm;
  float s = 0.f;
#pragma unroll
  for (int j = 1; j < K1n; ++j) {
    const int c = idx2[(b * Mm + m) * K1n + j];
    s += A[((size_t)(b * Nn + n)) * Mm + c];
  }
  refined[i] = expf(margin - s * (1.0f / (K1n - 1))) * dist[i];
}

// ---------------- src_corr[b,c,n] = sum_m tgt[b,c,m]*matching[b,n,m] -------
__global__ __launch_bounds__(256) void kn_src_corr(const float* __restrict__ tgt,
                                                   const float* __restrict__ matching,
                                                   float* __restrict__ src_corr) {
  const int row = blockIdx.x;             // b*N + n
  const int tid = threadIdx.x;
  const int b = row / Nn, n = row % Nn;
  __shared__ float s0[256], s1[256], s2[256];
  float a0 = 0.f, a1 = 0.f, a2 = 0.f;
  const size_t base = (size_t)row * Mm;
  for (int m = tid; m < Mm; m += 256) {
    const float mt = matching[base + m];
    a0 += tgt[(b*3 + 0)*Mm + m] * mt;
    a1 += tgt[(b*3 + 1)*Mm + m] * mt;
    a2 += tgt[(b*3 + 2)*Mm + m] * mt;
  }
  s0[tid] = a0; s1[tid] = a1; s2[tid] = a2; __syncthreads();
  for (int s = 128; s > 0; s >>= 1) {
    if (tid < s) { s0[tid]+=s0[tid+s]; s1[tid]+=s1[tid+s]; s2[tid]+=s2[tid+s]; }
    __syncthreads();
  }
  if (tid == 0) {
    src_corr[(b*3 + 0)*Nn + n] = s0[0];
    src_corr[(b*3 + 1)*Nn + n] = s1[0];
    src_corr[(b*3 + 2)*Nn + n] = s2[0];
  }
}

// --------------------------- discriminator MLP -----------------------------
__global__ __launch_bounds__(128) void kn_disc(const float* __restrict__ src,
                                               const float* __restrict__ src_corr,
                                               const float* __restrict__ src_knn,
                                               const int* __restrict__ src_idx,
                                               const float* __restrict__ w1, const float* __restrict__ b1,
                                               const float* __restrict__ w2, const float* __restrict__ b2,
                                               const float* __restrict__ w3, const float* __restrict__ b3,
                                               const float* __restrict__ w4, const float* __restrict__ b4,
                                               float* __restrict__ weight) {
  __shared__ float sW1[64*6], sB1[64], sW2[64*64], sB2[64], sW3[32*64], sB3[32], sW4[32], sB4[1];
  const int tid = threadIdx.x;
  for (int i = tid; i < 384;  i += 128) sW1[i] = w1[i];
  for (int i = tid; i < 64;   i += 128) { sB1[i] = b1[i]; sB2[i] = b2[i]; }
  for (int i = tid; i < 4096; i += 128) sW2[i] = w2[i];
  for (int i = tid; i < 2048; i += 128) sW3[i] = w3[i];
  for (int i = tid; i < 32;   i += 128) { sB3[i] = b3[i]; sW4[i] = w4[i]; }
  if (tid == 0) sB4[0] = b4[0];
  __syncthreads();

  const int p = blockIdx.x * 128 + tid;   // point id in [0, B*N)  (exact fit)
  const int b = p / Nn, n = p % Nn;
  float pc[3], ps[3];
#pragma unroll
  for (int c = 0; c < 3; ++c) {
    pc[c] = src_corr[(b*3 + c)*Nn + n];
    ps[c] = src[(b*3 + c)*Nn + n];
  }
  float g[64];
#pragma unroll
  for (int o = 0; o < 64; ++o) g[o] = -3.4e38f;

  for (int kk = 0; kk < Kk; ++kk) {
    const int nn = src_idx[p * Kk + kk];
    const int bb = nn / Nn, n2 = nn % Nn;
    float x[6];
#pragma unroll
    for (int c = 0; c < 3; ++c) x[c] = pc[c] - src_corr[(bb*3 + c)*Nn + n2];
    const float* skn = src_knn + ((size_t)p * Kk + kk) * 3;
#pragma unroll
    for (int c = 0; c < 3; ++c) x[3 + c] = ps[c] - skn[c];

    float h1[64];
#pragma unroll
    for (int o = 0; o < 64; ++o) {
      float s = sB1[o];
#pragma unroll
      for (int c = 0; c < 6; ++c) s += sW1[o*6 + c] * x[c];
      h1[o] = fmaxf(s, 0.f);
    }
    for (int pp = 0; pp < 64; ++pp) {
      float s = sB2[pp];
#pragma unroll
      for (int o = 0; o < 64; ++o) s += sW2[pp*64 + o] * h1[o];
      s = fmaxf(s, 0.f);
      g[pp] = fmaxf(g[pp], s);
    }
  }
  float g3[32];
#pragma unroll
  for (int q = 0; q < 32; ++q) {
    float s = sB3[q];
#pragma unroll
    for (int pp = 0; pp < 64; ++pp) s += sW3[q*64 + pp] * g[pp];
    g3[q] = fmaxf(s, 0.f);
  }
  float s = sB4[0];
#pragma unroll
  for (int q = 0; q < 32; ++q) s += sW4[q] * g3[q];
  weight[p] = 1.0f / (1.0f + expf(-s));
}

// --------------------- weighted means + covariance H -----------------------
__device__ __forceinline__ float blk_reduce(float v, float* red, int tid) {
  __syncthreads();
  red[tid] = v; __syncthreads();
  for (int s = 128; s > 0; s >>= 1) {
    if (tid < s) red[tid] += red[tid + s];
    __syncthreads();
  }
  float r = red[0]; __syncthreads();
  return r;
}

__global__ __launch_bounds__(256) void kn_rigid_pre(const float* __restrict__ src,
                                                    const float* __restrict__ src_corr,
                                                    const float* __restrict__ weight,
                                                    float* __restrict__ hm) {
  const int b = blockIdx.x, tid = threadIdx.x;
  __shared__ float red[256];
  float s = 0.f;
  for (int n = tid; n < Nn; n += 256) s += weight[b*Nn + n];
  const float inv = 1.0f / (blk_reduce(s, red, tid) + 1e-8f);

  float ms[3], mc[3];
  for (int c = 0; c < 3; ++c) {
    float a = 0.f, v = 0.f;
    for (int n = tid; n < Nn; n += 256) {
      const float w = weight[b*Nn + n] * inv;
      a += src[(b*3 + c)*Nn + n] * w;
      v += src_corr[(b*3 + c)*Nn + n] * w;
    }
    ms[c] = blk_reduce(a, red, tid);
    mc[c] = blk_reduce(v, red, tid);
  }
  for (int i = 0; i < 3; ++i)
    for (int j = 0; j < 3; ++j) {
      float a = 0.f;
      for (int n = tid; n < Nn; n += 256) {
        const float w = weight[b*Nn + n] * inv;
        a += (src[(b*3 + i)*Nn + n] - ms[i]) * w *
             (src_corr[(b*3 + j)*Nn + n] - mc[j]);
      }
      const float H = blk_reduce(a, red, tid);
      if (tid == 0) hm[b*16 + i*3 + j] = H;
    }
  if (tid == 0)
    for (int c = 0; c < 3; ++c) { hm[b*16 + 9 + c] = ms[c]; hm[b*16 + 12 + c] = mc[c]; }
}

// ---------------------- 3x3 SVD -> R,t (one thread/batch) ------------------
__device__ __forceinline__ float det3(const float M[3][3]) {
  return M[0][0]*(M[1][1]*M[2][2]-M[1][2]*M[2][1])
       - M[0][1]*(M[1][0]*M[2][2]-M[1][2]*M[2][0])
       + M[0][2]*(M[1][0]*M[2][1]-M[1][1]*M[2][0]);
}

__global__ void kn_svd_rt(const float* __restrict__ hm, float* __restrict__ out) {
  if (threadIdx.x == 0) out[OUT_LOSS] = 0.f;   // reset loss each call
  const int b = threadIdx.x;
  if (b >= Bh) return;

  float H[3][3], sm[3], cm[3];
  for (int i = 0; i < 3; ++i)
    for (int j = 0; j < 3; ++j) H[i][j] = hm[b*16 + i*3 + j];
  for (int c = 0; c < 3; ++c) { sm[c] = hm[b*16 + 9 + c]; cm[c] = hm[b*16 + 12 + c]; }

  float A[3][3];
  for (int i = 0; i < 3; ++i)
    for (int j = 0; j < 3; ++j) {
      float a = 0.f;
      for (int k = 0; k < 3; ++k) a += H[k][i]*H[k][j];
      A[i][j] = a;
    }
  float V[3][3] = {{1,0,0},{0,1,0},{0,0,1}};
  for (int sweep = 0; sweep < 16; ++sweep) {
    for (int pi = 0; pi < 3; ++pi) {
      const int p = (pi == 0) ? 0 : (pi == 1) ? 0 : 1;
      const int q = (pi == 0) ? 1 : (pi == 1) ? 2 : 2;
      const float apq = A[p][q];
      if (fabsf(apq) < 1e-13f) continue;
      const float tau = (A[q][q] - A[p][p]) / (2.0f * apq);
      const float t = (tau >= 0.f ? 1.f : -1.f) / (fabsf(tau) + sqrtf(1.f + tau*tau));
      const float c = 1.0f / sqrtf(1.f + t*t);
      const float s = t * c;
      for (int k = 0; k < 3; ++k) {
        const float akp = A[k][p], akq = A[k][q];
        A[k][p] = c*akp - s*akq; A[k][q] = s*akp + c*akq;
      }
      for (int k = 0; k < 3; ++k) {
        const float apk = A[p][k], aqk = A[q][k];
        A[p][k] = c*apk - s*aqk; A[q][k] = s*apk + c*aqk;
      }
      for (int k = 0; k < 3; ++k) {
        const float vkp = V[k][p], vkq = V[k][q];
        V[k][p] = c*vkp - s*vkq; V[k][q] = s*vkp + c*vkq;
      }
    }
  }
  float lam[3] = {A[0][0], A[1][1], A[2][2]};
  for (int i = 0; i < 2; ++i)
    for (int j = 0; j < 2 - i; ++j)
      if (lam[j] < lam[j+1]) {
        float tl = lam[j]; lam[j] = lam[j+1]; lam[j+1] = tl;
        for (int k = 0; k < 3; ++k) { float tv = V[k][j]; V[k][j] = V[k][j+1]; V[k][j+1] = tv; }
      }
  float S[3], U[3][3];
  for (int i = 0; i < 3; ++i) S[i] = sqrtf(fmaxf(lam[i], 0.f));
  for (int i = 0; i < 3; ++i) {
    float u[3] = {0.f, 0.f, 0.f};
    for (int j = 0; j < 3; ++j)
      for (int r = 0; r < 3; ++r) u[r] += H[r][j] * V[j][i];
    const float inv = 1.0f / fmaxf(S[i], 1e-12f);
    for (int r = 0; r < 3; ++r) U[r][i] = u[r] * inv;
  }
  if (S[2] < 1e-6f * fmaxf(S[0], 1.f)) {  // degenerate: col2 = col0 x col1
    U[0][2] = U[1][0]*U[2][1] - U[2][0]*U[1][1];
    U[1][2] = U[2][0]*U[0][1] - U[0][0]*U[2][1];
    U[2][2] = U[0][0]*U[1][1] - U[1][0]*U[0][1];
  }
  const float d = det3(U) * det3(V);
  float R[3][3];
  for (int i = 0; i < 3; ++i)
    for (int k = 0; k < 3; ++k)
      R[i][k] = V[i][0]*U[k][0] + V[i][1]*U[k][1] + d*V[i][2]*U[k][2];
  for (int i = 0; i < 3; ++i) {
    float t = cm[i];
    for (int j = 0; j < 3; ++j) t -= R[i][j] * sm[j];
    out[OUT_T + b*3 + i] = t;
    for (int k = 0; k < 3; ++k) out[OUT_R + b*9 + i*3 + k] = R[i][k];
  }
}

// ------------------------------- top-k -------------------------------------
__global__ __launch_bounds__(256) void kn_topk(const float* __restrict__ weight,
                                               int* __restrict__ topk) {
  const int b = blockIdx.x, tid = threadIdx.x;
  __shared__ float wrow[Nn];
  for (int n = tid; n < Nn; n += 256) wrow[n] = weight[b*Nn + n];
  __syncthreads();
  for (int n = tid; n < Nn; n += 256) {
    const float wv = wrow[n];
    int rank = 0;
    for (int n2 = 0; n2 < Nn; ++n2) {
      const float o = wrow[n2];
      rank += (o > wv) || (o == wv && n2 < n);
    }
    if (rank < NKP) topk[b*NKP + rank] = n;
  }
}

// ------------------------- keypoint / knn outputs --------------------------
__global__ __launch_bounds__(256) void kn_outputs(const float* __restrict__ src,
                                                  const float* __restrict__ src_corr,
                                                  const int* __restrict__ src_idx,
                                                  const int* __restrict__ topk,
                                                  float* __restrict__ out) {
  const int gid = blockIdx.x * blockDim.x + threadIdx.x;
  if (gid >= Bh * NKP) return;
  const int b = gid / NKP, i = gid % NKP;
  const int n = topk[gid];

  float R[3][3], t[3];
  for (int r = 0; r < 3; ++r) {
    t[r] = out[OUT_T + b*3 + r];
    for (int c = 0; c < 3; ++c) R[r][c] = out[OUT_R + b*9 + r*3 + c];
  }
  float sp[3], cp[3];
  for (int c = 0; c < 3; ++c) {
    sp[c] = src[(b*3 + c)*Nn + n];
    cp[c] = src_corr[(b*3 + c)*Nn + n];
    out[OUT_SKP + (b*3 + c)*NKP + i] = sp[c];
    out[OUT_TKP + (b*3 + c)*NKP + i] = cp[c];
  }
  float tp[3];
  for (int r = 0; r < 3; ++r)
    tp[r] = R[r][0]*sp[0] + R[r][1]*sp[1] + R[r][2]*sp[2] + t[r];

  for (int kk = 0; kk < Kk; ++kk) {
    const int nn = src_idx[(b*Nn + n)*Kk + kk];
    const int bb = nn / Nn, n2 = nn % Nn;
    float qs[3], qc[3];
    for (int c = 0; c < 3; ++c) {
      qs[c] = src[(bb*3 + c)*Nn + n2];
      qc[c] = src_corr[(bb*3 + c)*Nn + n2];
    }
    float tq[3];
    for (int r = 0; r < 3; ++r)
      tq[r] = R[r][0]*qs[0] + R[r][1]*qs[1] + R[r][2]*qs[2] + t[r];
    for (int c = 0; c < 3; ++c) {
      out[OUT_TKNN + ((b*3 + c)*NKP + i)*Kk + kk] = cp[c] - qc[c];
      out[OUT_SKNN + ((b*3 + c)*NKP + i)*Kk + kk] = tp[c] - tq[c];
    }
  }
}

// ---------------------------------- loss -----------------------------------
__global__ __launch_bounds__(256) void kn_loss(const float* __restrict__ matching,
                                               const int* __restrict__ topk,
                                               const int* __restrict__ argm,
                                               float* __restrict__ out) {
  const int gid = blockIdx.x * blockDim.x + threadIdx.x;
  if (gid >= Bh * NKP) return;
  const int b = gid / NKP;
  const int n = topk[gid];
  const int am = argm[b*Nn + n];
  const float v = matching[((size_t)(b*Nn + n)) * Mm + am];
  atomicAdd(&out[OUT_LOSS], -logf(v + 1e-15f) * (1.0f / (Bh * NKP)));
}

// ------------------------------- launcher ----------------------------------
extern "C" void kernel_launch(void* const* d_in, const int* in_sizes, int n_in,
                              void* d_out, int out_size, void* d_ws, size_t ws_size,
                              hipStream_t stream) {
  const float* src  = (const float*)d_in[0];
  const float* tgt  = (const float*)d_in[1];
  const float* xe   = (const float*)d_in[2];
  const float* ye   = (const float*)d_in[3];
  const int*   sidx = (const int*)d_in[4];
  const float* sknn = (const float*)d_in[6];
  const int*   sidx1= (const int*)d_in[9];
  const int*   idx2 = (const int*)d_in[10];
  const float* w1 = (const float*)d_in[12]; const float* b1 = (const float*)d_in[13];
  const float* w2 = (const float*)d_in[14]; const float* b2 = (const float*)d_in[15];
  const float* w3 = (const float*)d_in[16]; const float* b3 = (const float*)d_in[17];
  const float* w4 = (const float*)d_in[18]; const float* b4 = (const float*)d_in[19];
  const float* margin = (const float*)d_in[21];

  float* out = (float*)d_out;
  float* ws  = (float*)d_ws;
  float* xx    = ws + WS_XX;
  float* yy    = ws + WS_YY;
  float* dist  = ws + WS_DIST;
  float* score = ws + WS_SCORE;   // scores -> refined -> matching (reused)
  float* Abuf  = ws + WS_A;
  float* corr  = ws + WS_CORR;
  float* wgt   = ws + WS_W;
  float* hm    = ws + WS_HM;
  int*   topk  = (int*)(ws + WS_TOPK);
  int*   argm  = (int*)(ws + WS_ARG);

  kn_norms<<<(Bh*(Nn+Mm) + 255)/256, 256, 0, stream>>>(xe, ye, xx, yy);
  kn_dist_wmma<<<dim3(Nn/32, Mm/64, Bh), 128, 0, stream>>>(xe, ye, xx, yy, dist);
  kn_row_softmax<<<Bh*Nn, 256, 0, stream>>>(dist, score, nullptr);           // scores
  kn_A<<<Bh*Nn, 256, 0, stream>>>(score, sidx1, Abuf);
  kn_refined<<<(BNM + 255)/256, 256, 0, stream>>>(Abuf, dist, idx2, margin, score); // refined
  kn_row_softmax<<<Bh*Nn, 256, 0, stream>>>(score, score, argm);             // matching
  kn_src_corr<<<Bh*Nn, 256, 0, stream>>>(tgt, score, corr);
  kn_disc<<<(Bh*Nn)/128, 128, 0, stream>>>(src, corr, sknn, sidx,
                                           w1, b1, w2, b2, w3, b3, w4, b4, wgt);
  kn_rigid_pre<<<Bh, 256, 0, stream>>>(src, corr, wgt, hm);
  kn_svd_rt<<<1, 32, 0, stream>>>(hm, out);
  kn_topk<<<Bh, 256, 0, stream>>>(wgt, topk);
  kn_outputs<<<(Bh*NKP + 255)/256, 256, 0, stream>>>(src, corr, sidx, topk, out);
  kn_loss<<<(Bh*NKP + 255)/256, 256, 0, stream>>>(score, topk, argm, out);
}
